// SameTimeFlashFusionMultiViewSelfAttention_89258010346066
// MI455X (gfx1250) — compile-verified, matrix operand order checked by reading, not run on hardware
//
#include <hip/hip_runtime.h>
#include <math.h>

// ---------------- problem constants ----------------
constexpr int B_  = 1;
constexpr int V_  = 2;
constexpr int F_  = 4;
constexpr int H_  = 16;
constexpr int W_  = 28;
constexpr int TPF = H_ * W_;        // 448
constexpr int VS  = F_ * TPF;       // 1792
constexpr int TS  = V_ * VS;        // 3584
constexpr int DIM = 1536;
constexpr int NH  = 12;
constexpr int HD  = DIM / NH;       // 128
constexpr float EPS   = 1e-6f;
constexpr float SCALE = 0.08838834764831845f;   // 1/sqrt(128)

typedef __attribute__((ext_vector_type(16))) __bf16 v16bf;
typedef __attribute__((ext_vector_type(8)))  float  v8f;
typedef __attribute__((ext_vector_type(4)))  unsigned v4u;
typedef __attribute__((ext_vector_type(8)))  int    v8i;
typedef __attribute__((ext_vector_type(4)))  int    v4i;

#if __has_builtin(__builtin_amdgcn_tensor_load_to_lds)
#define HAS_TDM 1
#else
#define HAS_TDM 0
#endif
#if __has_builtin(__builtin_amdgcn_global_load_async_to_lds_b128)
#define HAS_ASYNC 1
#else
#define HAS_ASYNC 0
#endif

// typed pointers for the async global->LDS builtin: b128 = v4i, AS1 src / AS3 dst
typedef __attribute__((address_space(1))) v4i as1_v4i;
typedef __attribute__((address_space(3))) v4i as3_v4i;

__device__ __forceinline__ as1_v4i* gbl_v4i(const void* p) {
  return (as1_v4i*)(size_t)p;
}
__device__ __forceinline__ as3_v4i* lds_v4i(void* p) {
  // ptr -> int -> addrspace(3) ptr pattern (compile-safe cast)
  return (as3_v4i*)(unsigned)(size_t)p;
}

#if HAS_TDM
// One TDM descriptor: 2D tensor of bf16 (data_size=2B), tile tileW x tileH,
// tensor bounds tensorW x tensorH, row stride strideW (all in elements).
__device__ __forceinline__ void tdm_load_tile_2d(const __bf16* gsrc, void* ldsDst,
                                                 int tensorW, int tensorH,
                                                 int tileW, int tileH, int strideW) {
  const unsigned long long ga = (unsigned long long)(size_t)gsrc;
  const unsigned ldsOff = (unsigned)(size_t)ldsDst;
  v4u g0;
  g0[0] = 1u;                                           // count=1 valid descriptor
  g0[1] = ldsOff;                                       // lds_addr
  g0[2] = (unsigned)(ga & 0xffffffffu);                 // global_addr[31:0]
  g0[3] = (unsigned)((ga >> 32) & 0x01ffffffu) | (2u << 30); // addr[56:32] | type=2
  v8i g1;
  g1[0] = (1 << 16);                                    // wg_mask=0, data_size=1 (2B)
  g1[1] = (tensorW & 0xffff) << 16;                     // tensor_dim0[15:0]
  g1[2] = ((tensorW >> 16) & 0xffff) | ((tensorH & 0xffff) << 16); // dim0 hi | dim1 lo
  g1[3] = ((tensorH >> 16) & 0xffff) | ((tileW & 0xffff) << 16);   // dim1 hi | tile_dim0
  g1[4] = (tileH & 0xffff);                             // tile_dim1 (tile_dim2 = 0)
  g1[5] = strideW;                                      // tensor_dim0_stride[31:0]
  g1[6] = 0;                                            // stride hi, dim1_stride lo
  g1[7] = 0;
  v4i gz; gz[0] = 0; gz[1] = 0; gz[2] = 0; gz[3] = 0;   // groups 2/3 unused (2D)
#if __clang_major__ >= 23
  v8i gz8; gz8[0]=0; gz8[1]=0; gz8[2]=0; gz8[3]=0; gz8[4]=0; gz8[5]=0; gz8[6]=0; gz8[7]=0;
  __builtin_amdgcn_tensor_load_to_lds(g0, g1, gz, gz, gz8, 0);
#else
  __builtin_amdgcn_tensor_load_to_lds(g0, g1, gz, gz, 0);
#endif
}
__device__ __forceinline__ void tdm_wait() {
#if __has_builtin(__builtin_amdgcn_s_wait_tensorcnt)
  __builtin_amdgcn_s_wait_tensorcnt(0);
#else
  asm volatile("s_wait_tensorcnt 0x0" ::: "memory");
#endif
}
#endif

// ---------------- cast f32 -> bf16 ----------------
__global__ __launch_bounds__(256)
void cast_f32_bf16_kernel(const float* __restrict__ src, __bf16* __restrict__ dst, int n) {
  int i = blockIdx.x * 256 + threadIdx.x;
  if (i < n) dst[i] = (__bf16)src[i];
}

// ---------------- WMMA GEMM: C[m,n] = sum_k A[m,k] * W[n,k] + bias[n] ----------------
// A: bf16 [M,K] row-major, W: bf16 [N,K] row-major (weight), C: f32 [M,N]
// block = 256 threads = 8 waves, block tile 128(M) x 64(N), waves 4x2, wave tile 32x32.
// A tile staged by the Tensor Data Mover, B tile by async global->LDS loads.
__global__ __launch_bounds__(256)
void gemm_bf16_kernel(const __bf16* __restrict__ A, const __bf16* __restrict__ Wt,
                      const float* __restrict__ bias, float* __restrict__ Cout,
                      int M, int N, int K) {
  __shared__ alignas(16) __bf16 sA[128][32];   // 8 KB
  __shared__ alignas(16) __bf16 sB[64][32];    // 4 KB

  const int t    = threadIdx.x;
  const int wave = t >> 5;
  const int lane = t & 31;
  const int wm   = wave >> 1;          // 0..3
  const int wn   = wave & 1;           // 0..1
  const int m0   = blockIdx.x * 128;
  const int n0   = blockIdx.y * 64;
  const int lm   = lane & 15;
  const int lh   = (lane >> 4) * 16;

  v8f acc[2][2] = {};

  for (int kk = 0; kk < K; kk += 32) {
    // ---- stage A tile 128x32 ----
#if HAS_TDM
    if (wave == 0) {
      // one DMA for the whole tile: tensor [M,K], tile 32(w) x 128(h), stride K
      tdm_load_tile_2d(A + (size_t)m0 * K + kk, &sA[0][0], K, M, 32, 128, K);
    }
#else
    {
      const int row = t >> 1, col = (t & 1) * 16;
      const uint4* gp = (const uint4*)(A + (size_t)(m0 + row) * K + kk + col);
      uint4* sp = (uint4*)(&sA[row][col]);
      sp[0] = gp[0];
      sp[1] = gp[1];
      if (kk + 32 < K)
        __builtin_prefetch(A + (size_t)(m0 + row) * K + kk + 32 + col, 0, 3);
    }
#endif
    // ---- stage W tile 64x32 ----
    {
      const int row = t >> 2, col = (t & 3) * 8;
      const __bf16* gp = Wt + (size_t)(n0 + row) * K + kk + col;
#if HAS_ASYNC
      __builtin_amdgcn_global_load_async_to_lds_b128(gbl_v4i(gp), lds_v4i(&sB[row][col]), 0, 0);
#else
      *(uint4*)(&sB[row][col]) = *(const uint4*)gp;
#endif
      if (kk + 32 < K)
        __builtin_prefetch(gp + 32, 0, 3);
    }
    // ---- waits + barrier ----
#if HAS_TDM
    if (wave == 0) tdm_wait();
#endif
#if HAS_ASYNC
    asm volatile("s_wait_asynccnt 0x0" ::: "memory");
#endif
    __syncthreads();

    v16bf af[2], bf[2];
    #pragma unroll
    for (int i = 0; i < 2; ++i)
      af[i] = *(const v16bf*)(&sA[wm * 32 + i * 16 + lm][lh]);
    #pragma unroll
    for (int j = 0; j < 2; ++j)
      bf[j] = *(const v16bf*)(&sB[wn * 32 + j * 16 + lm][lh]);

    #pragma unroll
    for (int i = 0; i < 2; ++i)
      #pragma unroll
      for (int j = 0; j < 2; ++j)
        acc[i][j] = __builtin_amdgcn_wmma_f32_16x16x32_bf16(
            false, af[i], false, bf[j], (short)0, acc[i][j], false, false);
    __syncthreads();
  }

  const int hf = lane >> 4;
  #pragma unroll
  for (int i = 0; i < 2; ++i)
    #pragma unroll
    for (int j = 0; j < 2; ++j) {
      const int n = n0 + wn * 32 + j * 16 + lm;
      const float bv = bias ? bias[n] : 0.0f;
      #pragma unroll
      for (int r = 0; r < 8; ++r) {
        const int m = m0 + wm * 32 + i * 16 + r + 8 * hf;
        Cout[(size_t)m * N + n] = acc[i][j][r] + bv;
      }
    }
}

// ---------------- RMSNorm + RoPE (q/k), f32 in -> bf16 out ----------------
__global__ __launch_bounds__(256)
void norm_rope_kernel(const float* __restrict__ lin, const float* __restrict__ normw,
                      const float* __restrict__ freqs, __bf16* __restrict__ outb) {
  __shared__ float srow[DIM];
  __shared__ float ssum[8];
  const int row = blockIdx.x;
  const float* src = lin + (size_t)row * DIM;

  float s = 0.f;
  for (int i = threadIdx.x; i < DIM; i += 256) {
    const float v = src[i];
    srow[i] = v;
    s += v * v;
  }
  #pragma unroll
  for (int m = 1; m < 32; m <<= 1) s += __shfl_xor(s, m, 32);
  if ((threadIdx.x & 31) == 0) ssum[threadIdx.x >> 5] = s;
  __syncthreads();
  float tot = 0.f;
  #pragma unroll
  for (int w = 0; w < 8; ++w) tot += ssum[w];
  const float rms = rsqrtf(tot / (float)DIM + EPS);

  const int pos = row % VS;
  const int fr  = pos / TPF;
  const int rem = pos % TPF;
  const int hh  = rem / W_;
  const int ww  = rem % W_;

  for (int p = threadIdx.x; p < NH * (HD / 2); p += 256) {
    const int h = p >> 6;
    const int i = p & 63;
    const int prow = (i < 22) ? fr : (i < 43) ? hh : ww;
    const float ang = freqs[prow * 64 + i];
    const float c = __cosf(ang), sn = __sinf(ang);
    const int base = h * HD + 2 * i;
    const float x0 = srow[base]     * rms * normw[base];
    const float x1 = srow[base + 1] * rms * normw[base + 1];
    outb[(size_t)row * DIM + base]     = (__bf16)(x0 * c - x1 * sn);
    outb[(size_t)row * DIM + base + 1] = (__bf16)(x0 * sn + x1 * c);
  }
}

// ---------------- V: cast + transpose to [DIM][TS] bf16 ----------------
__global__ __launch_bounds__(256)
void v_transpose_kernel(const float* __restrict__ vlin, __bf16* __restrict__ vT) {
  const size_t idx = (size_t)blockIdx.x * 256 + threadIdx.x;
  if (idx >= (size_t)TS * DIM) return;
  const int row = (int)(idx / DIM);
  const int d   = (int)(idx % DIM);
  vT[(size_t)d * TS + row] = (__bf16)vlin[idx];
}

// ---------------- flash attention (one wave per 16-query tile) ----------------
__global__ __launch_bounds__(32)
void attn_kernel(const __bf16* __restrict__ qb, const __bf16* __restrict__ kb,
                 const __bf16* __restrict__ vT, float* __restrict__ outp,
                 float* __restrict__ lse, int L, int Fch, int swap) {
  __shared__ alignas(64) __bf16 sP[16 * 32];

  const int qt   = blockIdx.x;
  const int h    = blockIdx.y;
  const int ch   = blockIdx.z;
  const int kch  = swap ? ((1 - ch / Fch) * Fch + (ch % Fch)) : ch;
  const int lane = threadIdx.x;
  const int lm   = lane & 15;
  const int lh   = (lane >> 4) * 16;
  const int hf   = lane >> 4;

  v16bf qf[4];
  const __bf16* qbase = qb + (size_t)(ch * L + qt * 16 + lm) * DIM + h * HD;
  #pragma unroll
  for (int kk = 0; kk < 4; ++kk)
    qf[kk] = *(const v16bf*)(qbase + kk * 32 + lh);

  v8f acc[8] = {};
  float mrow[8], rsum[8];
  #pragma unroll
  for (int r = 0; r < 8; ++r) { mrow[r] = -1e30f; rsum[r] = 0.f; }

  for (int ks = 0; ks < L; ks += 32) {
    v8f s[2] = {};
    #pragma unroll
    for (int nt = 0; nt < 2; ++nt) {
      const __bf16* kbase = kb + (size_t)(kch * L + ks + nt * 16 + lm) * DIM + h * HD;
      #pragma unroll
      for (int kk = 0; kk < 4; ++kk) {
        const v16bf kf = *(const v16bf*)(kbase + kk * 32 + lh);
        s[nt] = __builtin_amdgcn_wmma_f32_16x16x32_bf16(
            false, qf[kk], false, kf, (short)0, s[nt], false, false);
      }
    }

    #pragma unroll
    for (int r = 0; r < 8; ++r) {
      const float a = s[0][r] * SCALE;
      const float b = s[1][r] * SCALE;
      s[0][r] = a; s[1][r] = b;
      float mx = fmaxf(a, b);
      mx = fmaxf(mx, __shfl_xor(mx, 1, 32));
      mx = fmaxf(mx, __shfl_xor(mx, 2, 32));
      mx = fmaxf(mx, __shfl_xor(mx, 4, 32));
      mx = fmaxf(mx, __shfl_xor(mx, 8, 32));
      const float nmax = fmaxf(mrow[r], mx);
      const float corr = __expf(mrow[r] - nmax);
      mrow[r] = nmax;
      #pragma unroll
      for (int dt = 0; dt < 8; ++dt) acc[dt][r] *= corr;
      const float p0 = __expf(a - nmax);
      const float p1 = __expf(b - nmax);
      s[0][r] = p0; s[1][r] = p1;
      float ps = p0 + p1;
      ps += __shfl_xor(ps, 1, 32);
      ps += __shfl_xor(ps, 2, 32);
      ps += __shfl_xor(ps, 4, 32);
      ps += __shfl_xor(ps, 8, 32);
      rsum[r] = rsum[r] * corr + ps;
    }

    #pragma unroll
    for (int nt = 0; nt < 2; ++nt)
      #pragma unroll
      for (int r = 0; r < 8; ++r) {
        const int m = r + 8 * hf;
        sP[m * 32 + nt * 16 + lm] = (__bf16)s[nt][r];
      }
    __syncthreads();
    const v16bf pf = *(const v16bf*)(&sP[lm * 32 + lh]);

    #pragma unroll
    for (int dt = 0; dt < 8; ++dt) {
      const __bf16* vbase = vT + (size_t)(h * HD + dt * 16 + lm) * TS + (kch * L + ks + lh);
      const v16bf vf = *(const v16bf*)vbase;
      acc[dt] = __builtin_amdgcn_wmma_f32_16x16x32_bf16(
          false, pf, false, vf, (short)0, acc[dt], false, false);
    }
    __syncthreads();
  }

  #pragma unroll
  for (int dt = 0; dt < 8; ++dt)
    #pragma unroll
    for (int r = 0; r < 8; ++r) {
      const int m = ch * L + qt * 16 + r + 8 * hf;
      outp[(size_t)m * DIM + h * HD + dt * 16 + lm] = acc[dt][r] / rsum[r];
    }
  if (lm == 0) {
    #pragma unroll
    for (int r = 0; r < 8; ++r) {
      const int m = ch * L + qt * 16 + r + 8 * hf;
      lse[(size_t)m * NH + h] = mrow[r] + __logf(rsum[r]);
    }
  }
}

// ---------------- LSE fusion -> bf16 ----------------
__global__ __launch_bounds__(256)
void fuse_kernel(const float* __restrict__ oi, const float* __restrict__ li,
                 const float* __restrict__ oc, const float* __restrict__ lc,
                 __bf16* __restrict__ fused) {
  const size_t idx = (size_t)blockIdx.x * 256 + threadIdx.x;
  if (idx >= (size_t)TS * DIM) return;
  const size_t row = idx / DIM;
  const int col = (int)(idx % DIM);
  const int h = col / HD;
  const float a = li[row * NH + h];
  const float b = lc[row * NH + h];
  const float mx = fmaxf(a, b);
  const float lu = mx + __logf(__expf(a - mx) + __expf(b - mx));
  const float wi = __expf(a - lu);
  const float wc = __expf(b - lu);
  fused[idx] = (__bf16)(wi * oi[idx] + wc * oc[idx]);
}

// ---------------- host launcher ----------------
extern "C" void kernel_launch(void* const* d_in, const int* in_sizes, int n_in,
                              void* d_out, int out_size, void* d_ws, size_t ws_size,
                              hipStream_t stream) {
  const float* x     = (const float*)d_in[0];
  const float* freqs = (const float*)d_in[1];
  const float* q_w   = (const float*)d_in[2];
  const float* q_b   = (const float*)d_in[3];
  const float* k_w   = (const float*)d_in[4];
  const float* k_b   = (const float*)d_in[5];
  const float* v_w   = (const float*)d_in[6];
  const float* v_b   = (const float*)d_in[7];
  const float* o_w   = (const float*)d_in[8];
  const float* o_b   = (const float*)d_in[9];
  const float* nq_w  = (const float*)d_in[10];
  const float* nk_w  = (const float*)d_in[11];
  float* out = (float*)d_out;

  char* ws = (char*)d_ws;
  size_t off = 0;
  auto take = [&](size_t bytes) -> char* {
    char* p = ws + off;
    off += (bytes + 255) & ~(size_t)255;
    return p;
  };

  const size_t NE = (size_t)TS * DIM;
  const size_t NW = (size_t)DIM * DIM;

  __bf16* xb   = (__bf16*)take(NE * 2);
  __bf16* wqb  = (__bf16*)take(NW * 2);
  __bf16* wkb  = (__bf16*)take(NW * 2);
  __bf16* wvb  = (__bf16*)take(NW * 2);
  __bf16* wob  = (__bf16*)take(NW * 2);
  float*  qlin = (float*)take(NE * 4);
  float*  klin = (float*)take(NE * 4);
  float*  vlin = (float*)take(NE * 4);
  __bf16* qbv  = (__bf16*)take(NE * 2);
  __bf16* kbv  = (__bf16*)take(NE * 2);
  __bf16* vT   = (__bf16*)take(NE * 2);
  float*  oi   = (float*)take(NE * 4);
  float*  li   = (float*)take((size_t)TS * NH * 4);
  float*  oc   = (float*)take(NE * 4);
  float*  lc   = (float*)take((size_t)TS * NH * 4);
  __bf16* fus  = (__bf16*)take(NE * 2);

  const int gElem  = (int)((NE + 255) / 256);
  const int gWElem = (int)((NW + 255) / 256);

  cast_f32_bf16_kernel<<<gElem,  256, 0, stream>>>(x,   xb,  (int)NE);
  cast_f32_bf16_kernel<<<gWElem, 256, 0, stream>>>(q_w, wqb, (int)NW);
  cast_f32_bf16_kernel<<<gWElem, 256, 0, stream>>>(k_w, wkb, (int)NW);
  cast_f32_bf16_kernel<<<gWElem, 256, 0, stream>>>(v_w, wvb, (int)NW);
  cast_f32_bf16_kernel<<<gWElem, 256, 0, stream>>>(o_w, wob, (int)NW);

  dim3 gGemm(TS / 128, DIM / 64);
  gemm_bf16_kernel<<<gGemm, 256, 0, stream>>>(xb, wqb, q_b, qlin, TS, DIM, DIM);
  gemm_bf16_kernel<<<gGemm, 256, 0, stream>>>(xb, wkb, k_b, klin, TS, DIM, DIM);
  gemm_bf16_kernel<<<gGemm, 256, 0, stream>>>(xb, wvb, v_b, vlin, TS, DIM, DIM);

  norm_rope_kernel<<<TS, 256, 0, stream>>>(qlin, nq_w, freqs, qbv);
  norm_rope_kernel<<<TS, 256, 0, stream>>>(klin, nk_w, freqs, kbv);
  v_transpose_kernel<<<gElem, 256, 0, stream>>>(vlin, vT);

  dim3 gAi(VS / 16, NH, B_ * V_);
  attn_kernel<<<gAi, 32, 0, stream>>>(qbv, kbv, vT, oi, li, VS, F_, 0);
  dim3 gAc(TPF / 16, NH, B_ * V_ * F_);
  attn_kernel<<<gAc, 32, 0, stream>>>(qbv, kbv, vT, oc, lc, TPF, F_, 1);

  fuse_kernel<<<gElem, 256, 0, stream>>>(oi, li, oc, lc, fus);
  gemm_bf16_kernel<<<gGemm, 256, 0, stream>>>(fus, wob, o_b, out, TS, DIM, DIM);

  (void)in_sizes; (void)n_in; (void)out_size; (void)ws_size;
}
